// VectorQuantizer_35055523070146
// MI455X (gfx1250) — matrix operand order repaired and checked by reading, not compiled, not run on hardware
//
#include <hip/hip_runtime.h>
#include <hip/hip_bf16.h>
#include <math.h>

// ---------------------------------------------------------------------------
// VQ-VAE vector quantizer for MI455X (gfx1250, wave32, WMMA).
//
// inputs : [16,1024,256] f32   (N = 16384 rows, D = 256)
// weight : [8192,256]    f32   (K = 8192 codes)
// outputs (concat, f32): quantized_st[N*256], loss[1], perplexity[1], codes[N]
//
// argmin_k ||x - w_k||^2  ==  argmin_k ( ||w_k||^2 - 2 x.w_k )   (drop ||x||^2)
// Matmul x.W^T done with v_wmma_f32_16x16x32_bf16 using hi/lo bf16 splits:
//   dot ~= xh.wh + xl.wh + xh.wl   (residual xl.wl ~ 1e-9, negligible)
// ---------------------------------------------------------------------------

typedef __attribute__((ext_vector_type(16))) __bf16 v16bf;
typedef __attribute__((ext_vector_type(8)))  __bf16 v8bf;
typedef __attribute__((ext_vector_type(8)))  float  v8f;

#define N_ROWS   16384
#define D_DIM    256
#define K_CODES  8192
#define N_ELEMS  (N_ROWS * D_DIM)      // 4194304

__device__ __forceinline__ unsigned short f32_to_bf16_rne(float f) {
  unsigned int u = __float_as_uint(f);
  unsigned int lsb = (u >> 16) & 1u;
  u += 0x7fffu + lsb;
  return (unsigned short)(u >> 16);
}
__device__ __forceinline__ float bf16_bits_to_f32(unsigned short h) {
  return __uint_as_float(((unsigned int)h) << 16);
}

// ---------------- prep: weight -> bf16 hi/lo + ||w||^2 ----------------------
__global__ __launch_bounds__(256) void vq_prep_w(const float* __restrict__ w,
                                                 unsigned short* __restrict__ whi,
                                                 unsigned short* __restrict__ wlo,
                                                 float* __restrict__ wnorm) {
  int c = blockIdx.x;            // code row
  int t = threadIdx.x;           // channel
  float v = w[c * D_DIM + t];
  unsigned short hb = f32_to_bf16_rne(v);
  float hf = bf16_bits_to_f32(hb);
  unsigned short lb = f32_to_bf16_rne(v - hf);
  whi[c * D_DIM + t] = hb;
  wlo[c * D_DIM + t] = lb;

  __shared__ float red[256];
  red[t] = v * v;
  __syncthreads();
  for (int s = 128; s > 0; s >>= 1) {
    if (t < s) red[t] += red[t + s];
    __syncthreads();
  }
  if (t == 0) wnorm[c] = red[0];
}

// ---------------- prep: inputs -> bf16 hi/lo --------------------------------
__global__ __launch_bounds__(256) void vq_prep_x(const float* __restrict__ x,
                                                 unsigned short* __restrict__ xhi,
                                                 unsigned short* __restrict__ xlo) {
  int i = blockIdx.x * 256 + threadIdx.x;
  float v = x[i];
  unsigned short hb = f32_to_bf16_rne(v);
  float hf = bf16_bits_to_f32(hb);
  xhi[i] = hb;
  xlo[i] = f32_to_bf16_rne(v - hf);
}

// ---------------- main: WMMA distance sweep + per-row argmin ----------------
// Block = 128 threads (4 waves). Wave w owns row-tile (blockIdx.x*4 + w):
// 16 input rows, sweeps all 512 code tiles of 16 codes, D in 8 k-chunks of 32.
__device__ __forceinline__ v16bf asm16(v8bf lo, v8bf hi) {
  v16bf r;
#pragma unroll
  for (int i = 0; i < 8; ++i) { r[i] = lo[i]; r[i + 8] = hi[i]; }
  return r;
}

__global__ __launch_bounds__(128) void vq_argmin(const __bf16* __restrict__ whi,
                                                 const __bf16* __restrict__ wlo,
                                                 const __bf16* __restrict__ xhi,
                                                 const __bf16* __restrict__ xlo,
                                                 const float* __restrict__ wnorm,
                                                 int* __restrict__ codes_out,
                                                 float* __restrict__ codes_f_out) {
  __shared__ __align__(16) float wn_lds[K_CODES];
  int tid = threadIdx.x;
  // Cooperative load of ||w||^2 into LDS (8192 f32 = 32 KB).
  {
    const float4* src = (const float4*)wnorm;
    float4* dst = (float4*)wn_lds;
    for (int i = tid; i < K_CODES / 4; i += 128) dst[i] = src[i];
  }
  __syncthreads();

  int wave = tid >> 5;
  int lane = tid & 31;
  int nrow = lane & 15;           // row within tile (B/C column)
  int half = (lane >> 4) & 1;     // lane half
  int rowtile = blockIdx.x * 4 + wave;
  int row = rowtile * 16 + nrow;

  // Resident B operand: x tile, hi+lo, 8 k-chunks. Lane holds 16 contiguous
  // bf16 channels per chunk: k = chunk*32 + half*16 + e  (e = 0..15).
  v16bf bxh[8], bxl[8];
  {
    const __bf16* xh_p = xhi + (size_t)row * D_DIM + half * 16;
    const __bf16* xl_p = xlo + (size_t)row * D_DIM + half * 16;
#pragma unroll
    for (int c = 0; c < 8; ++c) {
      bxh[c] = *(const v16bf*)(xh_p + c * 32);
      bxl[c] = *(const v16bf*)(xl_p + c * 32);
    }
  }

  // A operand addressing: lane -> code m = lane&15; channels match B:
  // elements 0..15 at k = chunk*32 + half*16 + e.
  int m = lane & 15;
  const __bf16* wh_lane = whi + (size_t)m * D_DIM + half * 16;
  const __bf16* wl_lane = wlo + (size_t)m * D_DIM + half * 16;

  float minval = 3.4e38f;
  int   minidx = 0;

  for (int ct = 0; ct < K_CODES / 16; ++ct) {
    const __bf16* whp = wh_lane + (size_t)ct * 16 * D_DIM;
    const __bf16* wlp = wl_lane + (size_t)ct * 16 * D_DIM;
    v8f c0 = {};   // xh.wh
    v8f c1 = {};   // xl.wh + xh.wl
#pragma unroll
    for (int c = 0; c < 8; ++c) {
      v16bf awh = asm16(*(const v8bf*)(whp + c * 32), *(const v8bf*)(whp + c * 32 + 8));
      v16bf awl = asm16(*(const v8bf*)(wlp + c * 32), *(const v8bf*)(wlp + c * 32 + 8));
      // D = A(16x32 codes) x B(32x16 rows) + C
      c0 = __builtin_amdgcn_wmma_f32_16x16x32_bf16(false, awh, false, bxh[c],
                                                   (short)0, c0, false, false);
      c1 = __builtin_amdgcn_wmma_f32_16x16x32_bf16(false, awh, false, bxl[c],
                                                   (short)0, c1, false, false);
      c1 = __builtin_amdgcn_wmma_f32_16x16x32_bf16(false, awl, false, bxh[c],
                                                   (short)0, c1, false, false);
    }
    // Scores: lane's VGPR v holds code (ct*16 + half*8 + v) for this row.
    int cbase = ct * 16 + half * 8;
    float4 wnA = *(const float4*)&wn_lds[cbase];
    float4 wnB = *(const float4*)&wn_lds[cbase + 4];
    float wn[8] = {wnA.x, wnA.y, wnA.z, wnA.w, wnB.x, wnB.y, wnB.z, wnB.w};
#pragma unroll
    for (int v = 0; v < 8; ++v) {
      float s = wn[v] - 2.0f * (c0[v] + c1[v]);
      if (s < minval) { minval = s; minidx = cbase + v; }
    }
  }

  // Merge the two lane-halves (codes m%16<8 vs >=8) for the same row.
  float oval = __shfl_xor(minval, 16, 32);
  int   oidx = __shfl_xor(minidx, 16, 32);
  if (oval < minval || (oval == minval && oidx < minidx)) {
    minval = oval; minidx = oidx;
  }
  if (lane < 16) {
    codes_out[row]  = minidx;
    codes_f_out[row] = (float)minidx;
  }
}

// ------- gather quantized + straight-through, per-row SSE, histogram --------
__global__ __launch_bounds__(256) void vq_gather(const float* __restrict__ x,
                                                 const float* __restrict__ w,
                                                 const int* __restrict__ codes,
                                                 float* __restrict__ outq,
                                                 float* __restrict__ partials,
                                                 int* __restrict__ counts) {
  int row = blockIdx.x;
  int ch = threadIdx.x;
  int code = codes[row];
  float q = w[(size_t)code * D_DIM + ch];
  float xi = x[(size_t)row * D_DIM + ch];
  outq[(size_t)row * D_DIM + ch] = xi + (q - xi);   // straight-through value
  float d = q - xi;
  __shared__ float red[256];
  red[ch] = d * d;
  __syncthreads();
  for (int s = 128; s > 0; s >>= 1) {
    if (ch < s) red[ch] += red[ch + s];
    __syncthreads();
  }
  if (ch == 0) {
    partials[row] = red[0];
    atomicAdd(&counts[code], 1);   // integer atomic: deterministic
  }
}

// ---------------- final: loss + perplexity (single block, fixed order) ------
__global__ __launch_bounds__(256) void vq_final(const float* __restrict__ partials,
                                                const int* __restrict__ counts,
                                                float* __restrict__ out_loss,
                                                float* __restrict__ out_perp) {
  int t = threadIdx.x;
  __shared__ float red[256];

  float acc = 0.0f;
  for (int i = t; i < N_ROWS; i += 256) acc += partials[i];
  red[t] = acc;
  __syncthreads();
  for (int s = 128; s > 0; s >>= 1) {
    if (t < s) red[t] += red[t + s];
    __syncthreads();
  }
  float sumsq = red[0];
  __syncthreads();

  float h = 0.0f;
  for (int k = t; k < K_CODES; k += 256) {
    float p = (float)counts[k] * (1.0f / (float)N_ROWS);
    h += p * logf(p + 1e-10f);
  }
  red[t] = h;
  __syncthreads();
  for (int s = 128; s > 0; s >>= 1) {
    if (t < s) red[t] += red[t + s];
    __syncthreads();
  }
  if (t == 0) {
    // loss = q_loss + 0.25*e_loss, both equal to MSE
    *out_loss = 1.25f * (sumsq / (float)N_ELEMS);
    *out_perp = expf(-red[0]);
  }
}

// ---------------------------------------------------------------------------
extern "C" void kernel_launch(void* const* d_in, const int* in_sizes, int n_in,
                              void* d_out, int out_size, void* d_ws, size_t ws_size,
                              hipStream_t stream) {
  const float* x = (const float*)d_in[0];   // [16,1024,256]
  const float* w = (const float*)d_in[1];   // [8192,256]

  float* out = (float*)d_out;
  float* out_q     = out;                       // 4194304
  float* out_loss  = out + N_ELEMS;             // 1
  float* out_perp  = out + N_ELEMS + 1;         // 1
  float* out_codes = out + N_ELEMS + 2;         // 16384

  // Workspace carve-up (256B aligned).
  char* ws = (char*)d_ws;
  size_t o = 0;
  auto carve = [&](size_t bytes) { void* p = ws + o; o += (bytes + 255) & ~(size_t)255; return p; };
  unsigned short* whi   = (unsigned short*)carve((size_t)K_CODES * D_DIM * 2);  // 4 MB
  unsigned short* wlo   = (unsigned short*)carve((size_t)K_CODES * D_DIM * 2);  // 4 MB
  unsigned short* xhi   = (unsigned short*)carve((size_t)N_ELEMS * 2);          // 8 MB
  unsigned short* xlo   = (unsigned short*)carve((size_t)N_ELEMS * 2);          // 8 MB
  float*          wnorm = (float*)carve((size_t)K_CODES * 4);
  int*            codes = (int*)carve((size_t)N_ROWS * 4);
  int*            counts= (int*)carve((size_t)K_CODES * 4);
  float*          parts = (float*)carve((size_t)N_ROWS * 4);

  hipMemsetAsync(counts, 0, (size_t)K_CODES * 4, stream);

  vq_prep_w<<<K_CODES, 256, 0, stream>>>(w, whi, wlo, wnorm);
  vq_prep_x<<<N_ELEMS / 256, 256, 0, stream>>>(x, xhi, xlo);

  // 1024 row-tiles, 4 waves/block -> 256 blocks of 128 threads.
  vq_argmin<<<N_ROWS / 64, 128, 0, stream>>>((const __bf16*)whi, (const __bf16*)wlo,
                                             (const __bf16*)xhi, (const __bf16*)xlo,
                                             wnorm, codes, out_codes);

  vq_gather<<<N_ROWS, 256, 0, stream>>>(x, w, codes, out_q, parts, counts);
  vq_final<<<1, 256, 0, stream>>>(parts, counts, out_loss, out_perp);
}
